// MS_NonLocal2d_38843684225268
// MI455X (gfx1250) — matrix operand order repaired and checked by reading, not compiled, not run on hardware
//
#include <hip/hip_runtime.h>

#define DIN   256
#define DD    128
#define NPOS  4096
#define BATCH 2
#define TQ    64
#define TK    64
#define NT    (NPOS / TK)

typedef __bf16 bf16_t;
typedef __attribute__((ext_vector_type(16))) __bf16 v16bf;
typedef __attribute__((ext_vector_type(8)))  __bf16 v8bf;
typedef __attribute__((ext_vector_type(8)))  float  v8f;
typedef __attribute__((ext_vector_type(4)))  int    v4i;

// ---- CDNA5 async global->LDS copy (ASYNCcnt) with sync fallback -----------
#if defined(__has_builtin)
#  if __has_builtin(__builtin_amdgcn_global_load_async_to_lds_b128)
#    define HAVE_ASYNC 1
#  endif
#endif
#ifndef HAVE_ASYNC
#  define HAVE_ASYNC 0
#endif

#if HAVE_ASYNC
typedef __attribute__((address_space(1))) v4i gv4i;   // global int4*
typedef __attribute__((address_space(3))) v4i lv4i;   // LDS int4*
static __device__ inline void async_cp16(const bf16_t* g, bf16_t* l) {
    __builtin_amdgcn_global_load_async_to_lds_b128((gv4i*)g, (lv4i*)l, 0, 0);
}
#  if __has_builtin(__builtin_amdgcn_s_wait_asynccnt)
#    define WAIT_ASYNC(n) __builtin_amdgcn_s_wait_asynccnt(n)
#  else
#    define WAIT_ASYNC(n) asm volatile("s_wait_asynccnt %0" ::"i"(n) : "memory")
#  endif
#endif

static __device__ inline v16bf cat8(v8bf lo, v8bf hi) {
    v16bf r;
#pragma unroll
    for (int i = 0; i < 8; ++i) { r[i] = lo[i]; r[8 + i] = hi[i]; }
    return r;
}

// ---------------------------------------------------------------------------
// Kernel 1: fused multi-scale projections.
//   Q[b,q,d]  = (x[.,q]·w_t1[d] + x2[.,q%1024]·w_t2[4d+q/1024]
//              + x4[.,q%256]·w_t3[16d+(q/256)%4] + biases) * 128^-0.5   (bf16, N-major)
//   K[b,q,d]  = x[.,q]·w_phi[d] + b_phi[d]                          (bf16, N-major)
//   V[b,d,q]  = x[.,q]·w_g[d]   + b_g[d]                            (bf16, D-major)
// ---------------------------------------------------------------------------
__global__ __launch_bounds__(256) void proj_kernel(
    const float* __restrict__ x,
    const float* __restrict__ w_t1, const float* __restrict__ b_t1,
    const float* __restrict__ w_t2, const float* __restrict__ b_t2,
    const float* __restrict__ w_t3, const float* __restrict__ b_t3,
    const float* __restrict__ w_phi, const float* __restrict__ b_phi,
    const float* __restrict__ w_g,  const float* __restrict__ b_g,
    bf16_t* __restrict__ Qo, bf16_t* __restrict__ Ko, bf16_t* __restrict__ Vo)
{
    __shared__ float cols[3][DIN][16];   // [scale-set][channel][pos]
    const int t   = threadIdx.x;
    const int bid = blockIdx.x;
    const int b   = bid >> 8;
    const int q0  = (bid & 255) << 4;
    const float* xb = x + (size_t)b * DIN * NPOS;

    {   // stage the three gathered x-columns for 16 positions (coalesced on hw)
        const int p  = t & 15, cb = t >> 4;
        const int hw1 = q0 + p;
        const int s2  = hw1 & 1023;
        const int hw2 = ((s2 >> 5) << 7) | ((s2 & 31) << 1);   // stride-2 sample
        const int s3  = hw1 & 255;
        const int hw3 = ((s3 >> 4) << 8) | ((s3 & 15) << 2);   // stride-4 sample
#pragma unroll
        for (int i = 0; i < 16; ++i) {
            const int c = cb + (i << 4);
            cols[0][c][p] = xb[(size_t)c * NPOS + hw1];
            cols[1][c][p] = xb[(size_t)c * NPOS + hw2];
            cols[2][c][p] = xb[(size_t)c * NPOS + hw3];
        }
    }
    __syncthreads();

    const int d  = t & 127;
    const int pg = t >> 7;                 // 2 position-groups of 8
    const int r2 = (d << 2) + (q0 >> 10);  // w_t2 row (constant across 16 pos)
    const int r3 = (d << 4) + ((q0 >> 8) & 3);

    float aQ[8], aK[8], aV[8];
#pragma unroll
    for (int j = 0; j < 8; ++j) {
        aQ[j] = b_t1[d] + b_t2[r2] + b_t3[r3];
        aK[j] = b_phi[d];
        aV[j] = b_g[d];
    }
    const float* wr1 = w_t1 + (size_t)d * DIN;
    const float* wr2 = w_t2 + (size_t)r2 * DIN;
    const float* wr3 = w_t3 + (size_t)r3 * DIN;
    const float* wrp = w_phi + (size_t)d * DIN;
    const float* wrg = w_g  + (size_t)d * DIN;

    for (int c = 0; c < DIN; ++c) {
        const float w1 = wr1[c], w2 = wr2[c], w3 = wr3[c], wp = wrp[c], wg = wrg[c];
        const float* c0 = &cols[0][c][pg * 8];
        const float* c1 = &cols[1][c][pg * 8];
        const float* c2 = &cols[2][c][pg * 8];
#pragma unroll
        for (int j = 0; j < 8; ++j) {
            const float v0 = c0[j], v1 = c1[j], v2 = c2[j];
            aQ[j] += w1 * v0 + w2 * v1 + w3 * v2;
            aK[j] += wp * v0;
            aV[j] += wg * v0;
        }
    }
    const float scale = 0.08838834764831845f;   // 128^-0.5 folded into Q
#pragma unroll
    for (int j = 0; j < 8; ++j) {
        const int q = q0 + pg * 8 + j;
        const size_t nd = ((size_t)b * NPOS + q) * DD + d;
        Qo[nd] = (bf16_t)(aQ[j] * scale);
        Ko[nd] = (bf16_t)aK[j];
        Vo[((size_t)b * DD + d) * NPOS + q] = (bf16_t)aV[j];
    }
}

// ---------------------------------------------------------------------------
// Kernel 2: flash-attention with v_wmma_f32_16x16x32_bf16, double-buffered
// async global->LDS staging (ASYNCcnt ping-pong), and explicitly software-
// pipelined LDS B-fragment loads (load frag f+1 before WMMA f).
// ---------------------------------------------------------------------------
__global__ __launch_bounds__(128) void attn_kernel(
    const bf16_t* __restrict__ Q, const bf16_t* __restrict__ K,
    const bf16_t* __restrict__ V, float* __restrict__ T)
{
    __shared__ __align__(32) bf16_t kt[2][TK * DD];
    __shared__ __align__(32) bf16_t vt[2][DD * TK];
    __shared__ __align__(32) bf16_t pbuf[4][16][72];   // per-wave P scratch (padded)

    const int t = threadIdx.x;
    const int wv = t >> 5, lane = t & 31;
    const int lhalf = lane >> 4, l15 = lane & 15;
    const int b  = blockIdx.x >> 6;
    const int q0 = (blockIdx.x & 63) << 6;

    const bf16_t* Kb = K + (size_t)b * NPOS * DD;
    const bf16_t* Vb = V + (size_t)b * DD * NPOS;

    // stage one 64-key tile into buffer bi (16 async b128 ops per wave)
    auto stage = [&](int kti, int bi) {
        const int k0 = kti * TK;
#pragma unroll
        for (int i = 0; i < 8; ++i) {
            const int u = i * 128 + t;
            const int key = u >> 4, sub = u & 15;
            const int ch = sub >> 1;
            bf16_t* dst = &kt[bi][key * 128 + ((ch ^ (key & 7)) << 4) + ((sub & 1) << 3)];
            const bf16_t* src = Kb + (size_t)(k0 + key) * DD + sub * 8;
#if HAVE_ASYNC
            async_cp16(src, dst);
#else
            *(v8bf*)dst = *(const v8bf*)src;
#endif
        }
#pragma unroll
        for (int i = 0; i < 8; ++i) {
            const int u = i * 128 + t;
            const int dd = u >> 3, sub = u & 7;
            const int ch = sub >> 1;
            bf16_t* dst = &vt[bi][dd * 64 + ((ch ^ (dd & 3)) << 4) + ((sub & 1) << 3)];
            const bf16_t* src = Vb + (size_t)dd * NPOS + k0 + sub * 8;
#if HAVE_ASYNC
            async_cp16(src, dst);
#else
            *(v8bf*)dst = *(const v8bf*)src;
#endif
        }
    };

    // Q A-fragments (16x32 bf16): lane<16 holds K{0..15} of row M=lane.
    v16bf qf[4];
    {
        const bf16_t* qrow = Q + ((size_t)b * NPOS + q0 + wv * 16 + l15) * DD;
#pragma unroll
        for (int dc = 0; dc < 4; ++dc) {
            const int base = dc * 32 + lhalf * 16;
            qf[dc] = cat8(*(const v8bf*)(qrow + base),
                          *(const v8bf*)(qrow + base + 8));
        }
    }
    v8f o[8];
#pragma unroll
    for (int i = 0; i < 8; ++i) o[i] = v8f{0, 0, 0, 0, 0, 0, 0, 0};
    float m_r[8], l_r[8];
#pragma unroll
    for (int i = 0; i < 8; ++i) { m_r[i] = -1e30f; l_r[i] = 0.f; }

    stage(0, 0);

    for (int kti = 0; kti < NT; ++kti) {
        const int cur = kti & 1;
        if (kti + 1 < NT) {
            stage(kti + 1, cur ^ 1);     // prefetch next tile into spare buffer
#if HAVE_ASYNC
            WAIT_ASYNC(16);              // in-order: tile kti landed, kti+1 in flight
#endif
        } else {
#if HAVE_ASYNC
            WAIT_ASYNC(0);
#endif
        }
        __syncthreads();                 // all waves' copies visible
        const bf16_t* ktc = kt[cur];
        const bf16_t* vtc = vt[cur];

        auto kaddr = [&](int c, int dc) {
            const int col = c * 16 + l15;
            const int ch  = dc * 2 + lhalf;
            return (const v16bf*)(ktc + col * 128 + ((ch ^ (col & 7)) << 4));
        };
        auto vaddr = [&](int kk, int oc) {
            const int dcol = oc * 16 + l15;
            const int ch   = kk * 2 + lhalf;
            return (const v16bf*)(vtc + dcol * 64 + ((ch ^ (dcol & 3)) << 4));
        };

        // ---- S = (Q*scale) K^T : 16 WMMA, B-frag loads pipelined 1 ahead ----
        v8f s[4];
        {
            v16bf bcur = *kaddr(0, 0);
#pragma unroll
            for (int f = 0; f < 16; ++f) {
                const int c = f >> 2, dc = f & 3;
                v16bf bnxt = bcur;
                if (f < 15) bnxt = *kaddr((f + 1) >> 2, (f + 1) & 3);
                v8f acc = (dc == 0) ? v8f{0, 0, 0, 0, 0, 0, 0, 0} : s[c];
                s[c] = __builtin_amdgcn_wmma_f32_16x16x32_bf16(
                           false, qf[dc], false, bcur, (short)0, acc, false, false);
                bcur = bnxt;
            }
        }

        // ---- online softmax: C-layout row = i + 8*lhalf, col = l15 ----
        float alpha[8];
#pragma unroll
        for (int i = 0; i < 8; ++i) {
            float mx = fmaxf(fmaxf(s[0][i], s[1][i]), fmaxf(s[2][i], s[3][i]));
#pragma unroll
            for (int off = 1; off < 16; off <<= 1)
                mx = fmaxf(mx, __shfl_xor(mx, off, 32));
            const float mn = fmaxf(m_r[i], mx);
            alpha[i] = __expf(m_r[i] - mn);
            m_r[i] = mn;
            float rs = 0.f;
#pragma unroll
            for (int c = 0; c < 4; ++c) {
                const float p = __expf(s[c][i] - mn);
                s[c][i] = p;
                rs += p;
            }
#pragma unroll
            for (int off = 1; off < 16; off <<= 1)
                rs += __shfl_xor(rs, off, 32);
            l_r[i] = l_r[i] * alpha[i] + rs;
        }
#pragma unroll
        for (int oc = 0; oc < 8; ++oc) {
#pragma unroll
            for (int i = 0; i < 8; ++i) o[oc][i] *= alpha[i];
        }
        // ---- P -> per-wave LDS (C layout -> A layout transpose) ----
#pragma unroll
        for (int c = 0; c < 4; ++c) {
#pragma unroll
            for (int i = 0; i < 8; ++i)
                pbuf[wv][lhalf * 8 + i][c * 16 + l15] = (bf16_t)s[c][i];
        }
        // ---- O += P V : 16 WMMA, V-frag loads pipelined 1 ahead ----
#pragma unroll
        for (int kk = 0; kk < 2; ++kk) {
            const bf16_t* pr = &pbuf[wv][l15][kk * 32 + lhalf * 16];
            const v16bf af = cat8(*(const v8bf*)pr, *(const v8bf*)(pr + 8));
            v16bf bcur = *vaddr(kk, 0);
#pragma unroll
            for (int oc = 0; oc < 8; ++oc) {
                v16bf bnxt = bcur;
                if (oc < 7) bnxt = *vaddr(kk, oc + 1);
                o[oc] = __builtin_amdgcn_wmma_f32_16x16x32_bf16(
                            false, af, false, bcur, (short)0, o[oc], false, false);
                bcur = bnxt;
            }
        }
        __syncthreads();                 // buf cur free for refill next iteration
    }
    // ---- normalize + write t as (N, D) fp32 ----
#pragma unroll
    for (int i = 0; i < 8; ++i) {
        const float inv = 1.0f / l_r[i];
        const int q = q0 + wv * 16 + lhalf * 8 + i;
        float* tp = T + ((size_t)b * NPOS + q) * DD + l15;
#pragma unroll
        for (int oc = 0; oc < 8; ++oc) tp[oc * 16] = o[oc][i] * inv;
    }
}

// ---------------------------------------------------------------------------
// Kernel 3: y = w_out (256x128) @ t^T + b_out + x   (WMMA bf16, fp32 accum)
// Block = 128 threads (4 waves): 64 out-channels x 64 positions.
// ---------------------------------------------------------------------------
__global__ __launch_bounds__(128) void out_kernel(
    const float* __restrict__ T, const float* __restrict__ w_out,
    const float* __restrict__ b_out, const float* __restrict__ x,
    float* __restrict__ y)
{
    const int t = threadIdx.x;
    const int wv = t >> 5, lane = t & 31;
    const int lhalf = lane >> 4, l15 = lane & 15;
    const int bid = blockIdx.x;
    const int b   = bid >> 8;
    const int r   = bid & 255;
    const int o0  = (r >> 6) << 6;
    const int hw0 = (r & 63) << 6;

    v16bf af[4];
    {
        const int orow = o0 + wv * 16 + l15;
        const float* wr = w_out + (size_t)orow * DD;
#pragma unroll
        for (int dc = 0; dc < 4; ++dc) {
            const int base = dc * 32 + lhalf * 16;
            v16bf a;
#pragma unroll
            for (int j = 0; j < 8; ++j) {
                a[j]     = (bf16_t)wr[base + j];
                a[8 + j] = (bf16_t)wr[base + 8 + j];
            }
            af[dc] = a;
        }
    }
    const float* Tb = T + (size_t)b * NPOS * DD;
#pragma unroll
    for (int c = 0; c < 4; ++c) {
        const int col = hw0 + c * 16 + l15;
        v8f acc = v8f{0, 0, 0, 0, 0, 0, 0, 0};
#pragma unroll
        for (int dc = 0; dc < 4; ++dc) {
            const float* tp = Tb + (size_t)col * DD + dc * 32 + lhalf * 16;
            v16bf bf;
#pragma unroll
            for (int j = 0; j < 16; ++j) bf[j] = (bf16_t)tp[j];
            acc = __builtin_amdgcn_wmma_f32_16x16x32_bf16(
                      false, af[dc], false, bf, (short)0, acc, false, false);
        }
#pragma unroll
        for (int i = 0; i < 8; ++i) {
            const int oo = o0 + wv * 16 + lhalf * 8 + i;
            const size_t idx = ((size_t)b * DIN + oo) * NPOS + col;
            y[idx] = acc[i] + b_out[oo] + x[idx];
        }
    }
}

// ---------------------------------------------------------------------------
extern "C" void kernel_launch(void* const* d_in, const int* in_sizes, int n_in,
                              void* d_out, int out_size, void* d_ws, size_t ws_size,
                              hipStream_t stream) {
    const float* x     = (const float*)d_in[0];
    const float* w_t1  = (const float*)d_in[1];
    const float* b_t1  = (const float*)d_in[2];
    const float* w_t2  = (const float*)d_in[3];
    const float* b_t2  = (const float*)d_in[4];
    const float* w_t3  = (const float*)d_in[5];
    const float* b_t3  = (const float*)d_in[6];
    const float* w_phi = (const float*)d_in[7];
    const float* b_phi = (const float*)d_in[8];
    const float* w_g   = (const float*)d_in[9];
    const float* b_g   = (const float*)d_in[10];
    const float* w_out = (const float*)d_in[11];
    const float* b_out = (const float*)d_in[12];
    float* y = (float*)d_out;

    const size_t nQ = (size_t)BATCH * NPOS * DD;
    bf16_t* Q = (bf16_t*)d_ws;
    bf16_t* K = Q + nQ;
    bf16_t* V = K + nQ;
    float*  T = (float*)(V + nQ);

    proj_kernel<<<BATCH * (NPOS / 16), 256, 0, stream>>>(
        x, w_t1, b_t1, w_t2, b_t2, w_t3, b_t3, w_phi, b_phi, w_g, b_g, Q, K, V);
    attn_kernel<<<BATCH * (NPOS / TQ), 128, 0, stream>>>(Q, K, V, T);
    out_kernel<<<BATCH * (DIN / 64) * (NPOS / 64), 128, 0, stream>>>(
        T, w_out, b_out, x, y);
}
